// TriModalCrossAttention_34497177321845
// MI455X (gfx1250) — matrix operand order repaired and checked by reading, not compiled, not run on hardware
//
#include <hip/hip_runtime.h>
#include <hip/hip_bf16.h>
#include <math.h>

typedef __attribute__((ext_vector_type(16))) _Float16 v16h;
typedef __attribute__((ext_vector_type(8)))  float    v8f;

// Problem constants
#define BATCH 2
#define CCH   256      // channels
#define NPOS  4096     // H*W
#define PDIM  64       // proj dim
#define NHEAD 4
#define HDIM  16
#define SCALE 0.25f    // HD^-0.5
#define GNEPS 1e-5f
#define LOG2E 1.44269504088896f

struct Params {
  const float* feat[3];
  const float* Wq[3]; const float* bq[3];
  const float* Wk[3]; const float* bk[3];
  const float* Wv[3]; const float* bv[3];
  const float* Wo[3]; const float* bo[3];
  const float* gam[3]; const float* bet[3];
  _Float16 *Qt, *Kt, *Vd, *O64;
  float *Y, *part, *stats;
  float* out;
};

#define WMMA_F16(A,B,C) __builtin_amdgcn_wmma_f32_16x16x32_f16(false,(A),false,(B),(short)0,(C),false,false)

// ---------------------------------------------------------------------------
// Kernel 1: Q/K/V projections.  One wave -> (proj, mod, batch, 16-col n-tile),
// computes all 64 p-rows (4 WMMA accumulators), K-chunks of 32 over channels.
// Qt/Kt stored [m][b][h][n][d] f16; V stored [m][b][h][d][n] f16.
// ---------------------------------------------------------------------------
__global__ __launch_bounds__(128) void proj_kernel(Params P) {
  const int lane = threadIdx.x & 31, wv = threadIdx.x >> 5;
  int widx = blockIdx.x * 4 + wv;              // [0, 4608)
  const int proj = widx / 1536; widx -= proj * 1536;
  const int m    = widx / 512;  widx -= m * 512;
  const int b    = widx / 256;
  const int nt   = widx & 255;
  const int nBase = nt * 16;
  const int hh = lane >> 4, l16 = lane & 15;

  const float *W, *bias, *src0, *src1;
  int cTot;
  const int o1 = (m == 0) ? 1 : 0;
  const int o2 = (m == 2) ? 1 : 2;
  if (proj == 0) {
    W = P.Wq[m]; bias = P.bq[m]; cTot = 256;
    src0 = P.feat[m] + (size_t)b * CCH * NPOS; src1 = src0;
  } else if (proj == 1) {
    W = P.Wk[m]; bias = P.bk[m]; cTot = 512;
    src0 = P.feat[o1] + (size_t)b * CCH * NPOS;
    src1 = P.feat[o2] + (size_t)b * CCH * NPOS;
  } else {
    W = P.Wv[m]; bias = P.bv[m]; cTot = 512;
    src0 = P.feat[o1] + (size_t)b * CCH * NPOS;
    src1 = P.feat[o2] + (size_t)b * CCH * NPOS;
  }

  v8f acc0 = {}, acc1 = {}, acc2 = {}, acc3 = {};
  for (int cb = 0; cb < cTot; cb += 32) {
    // B fragment: X tile (32 c x 16 n).  lane -> col n, elem e -> K = hh*16+e
    v16h bf;
#pragma unroll
    for (int e = 0; e < 16; e++) {
      const int c = cb + (hh << 4) + e;
      const float* s = (c < 256) ? src0 : src1;
      bf[e] = (_Float16)s[(size_t)(c & 255) * NPOS + nBase + l16];
    }
    // A fragments: W tiles (16 p x 32 c). lane -> row p, elem e -> K per layout
#pragma unroll
    for (int pt = 0; pt < 4; pt++) {
      v16h af;
      const int p = pt * 16 + l16;
#pragma unroll
      for (int e = 0; e < 16; e++) {
        const int c = cb + (hh << 4) + e;
        af[e] = (_Float16)W[p * cTot + c];
      }
      if      (pt == 0) acc0 = WMMA_F16(af, bf, acc0);
      else if (pt == 1) acc1 = WMMA_F16(af, bf, acc1);
      else if (pt == 2) acc2 = WMMA_F16(af, bf, acc2);
      else              acc3 = WMMA_F16(af, bf, acc3);
    }
  }

  // D layout: vgpr r, lane -> row M = r+8*hh (p local), col N = l16 (n local)
#pragma unroll
  for (int pt = 0; pt < 4; pt++) {
    v8f acc = (pt == 0) ? acc0 : (pt == 1) ? acc1 : (pt == 2) ? acc2 : acc3;
#pragma unroll
    for (int r = 0; r < 8; r++) {
      const int pl = r + (hh << 3);
      const float v = acc[r] + bias[pt * 16 + pl];
      const int n = nBase + l16;
      const size_t hb = (size_t)((m * 2 + b) * 4 + pt);
      if (proj == 2) {
        P.Vd[(hb * HDIM + pl) * NPOS + n] = (_Float16)v;
      } else {
        _Float16* dst = (proj == 0) ? P.Qt : P.Kt;
        dst[(hb * NPOS + n) * HDIM + pl] = (_Float16)v;
      }
    }
  }
}

// ---------------------------------------------------------------------------
// Kernel 2: flash attention.  One wave -> (mod, batch, head, 16-query tile).
// 64-key chunks: 4 QK WMMAs (A = Q zero-padded K=16->32) + 2 full-K PV WMMAs.
// Online softmax in log2 domain (v_exp_f32 native), shfl_xor row reductions
// inside 16-lane halves (matching the C/D layout M = r + 8*(lane>=16)).
// ---------------------------------------------------------------------------
__global__ __launch_bounds__(128) void attn_kernel(Params P) {
  __shared__ _Float16 plds[4][16][64];
  const int lane = threadIdx.x & 31, wv = threadIdx.x >> 5;
  int widx = blockIdx.x * 4 + wv;              // [0, 6144)
  const int m = widx / 2048; widx -= m * 2048;
  const int b = widx >> 10;  widx &= 1023;
  const int h = widx >> 8;
  const int qt = widx & 255;
  const int hh = lane >> 4, l16 = lane & 15;
  const size_t head = (size_t)((m * 2 + b) * 4 + h);
  const _Float16* Qh = P.Qt + head * NPOS * HDIM;
  const _Float16* Kh = P.Kt + head * NPOS * HDIM;
  const _Float16* Vh = P.Vd + head * HDIM * NPOS;
  const int qBase = qt * 16;
  const float C1 = SCALE * LOG2E;              // fold scale into log2-domain

  // Q A-fragment: lanes 0..15 elems 0..15 = q[d = e] for row n, lanes 16..31 = 0 (K pad)
  v16h aq = {};
#pragma unroll
  for (int e = 0; e < 16; e++)
    aq[e] = hh ? (_Float16)0 : Qh[(size_t)(qBase + l16) * HDIM + e];

  float mi[8], li[8];
  v8f o = {};
#pragma unroll
  for (int r = 0; r < 8; r++) { mi[r] = -__builtin_inff(); li[r] = 0.f; }

  for (int kb = 0; kb < NPOS; kb += 64) {
    // K B-fragments: contiguous 32B per lane from Kt[pos][d]
    v16h bk0 = *(const v16h*)(Kh + (size_t)(kb      + l16) * HDIM);
    v16h bk1 = *(const v16h*)(Kh + (size_t)(kb + 16 + l16) * HDIM);
    v16h bk2 = *(const v16h*)(Kh + (size_t)(kb + 32 + l16) * HDIM);
    v16h bk3 = *(const v16h*)(Kh + (size_t)(kb + 48 + l16) * HDIM);
    // V B-fragments: contiguous from Vd[d][kb + ...]
    v16h bv0 = *(const v16h*)(Vh + (size_t)l16 * NPOS + kb      + (hh << 4));
    v16h bv1 = *(const v16h*)(Vh + (size_t)l16 * NPOS + kb + 32 + (hh << 4));
    // prefetch next chunk while this one computes (global_prefetch_b8)
    __builtin_prefetch(Kh + (size_t)(kb + 64 + l16) * HDIM, 0, 1);
    __builtin_prefetch(Vh + (size_t)l16 * NPOS + kb + 64 + (hh << 4), 0, 1);

    v8f z = {};
    v8f s0 = WMMA_F16(aq, bk0, z);
    v8f s1 = WMMA_F16(aq, bk1, z);
    v8f s2 = WMMA_F16(aq, bk2, z);
    v8f s3 = WMMA_F16(aq, bk3, z);

#pragma unroll
    for (int r = 0; r < 8; r++) {
      const float a0 = s0[r] * C1, a1 = s1[r] * C1;
      const float a2 = s2[r] * C1, a3 = s3[r] * C1;
      float mx = fmaxf(fmaxf(a0, a1), fmaxf(a2, a3));
#pragma unroll
      for (int off = 1; off < 16; off <<= 1) mx = fmaxf(mx, __shfl_xor(mx, off, 32));
      const float mn = fmaxf(mi[r], mx);
      const float corr = __builtin_amdgcn_exp2f(mi[r] - mn);
      const float p0 = __builtin_amdgcn_exp2f(a0 - mn);
      const float p1 = __builtin_amdgcn_exp2f(a1 - mn);
      const float p2 = __builtin_amdgcn_exp2f(a2 - mn);
      const float p3 = __builtin_amdgcn_exp2f(a3 - mn);
      float rs = (p0 + p1) + (p2 + p3);
#pragma unroll
      for (int off = 1; off < 16; off <<= 1) rs += __shfl_xor(rs, off, 32);
      li[r] = li[r] * corr + rs;
      o[r] *= corr;
      mi[r] = mn;
      // stash probabilities row-major [n_local][m_local] for D->A layout change
      plds[wv][r + (hh << 3)][l16]      = (_Float16)p0;
      plds[wv][r + (hh << 3)][l16 + 16] = (_Float16)p1;
      plds[wv][r + (hh << 3)][l16 + 32] = (_Float16)p2;
      plds[wv][r + (hh << 3)][l16 + 48] = (_Float16)p3;
    }
    // P A-fragments (same-wave DS ordering guarantees store->load visibility)
    v16h ap0, ap1;
#pragma unroll
    for (int e = 0; e < 16; e++) {
      const int col = (hh << 4) + e;
      ap0[e] = plds[wv][l16][col];
      ap1[e] = plds[wv][l16][col + 32];
    }
    o = WMMA_F16(ap0, bv0, o);
    o = WMMA_F16(ap1, bv1, o);
  }

  // O in D layout: row n_local = r+8*hh, col d = l16.  Store [n][p] f16.
#pragma unroll
  for (int r = 0; r < 8; r++) {
    const float val = o[r] / li[r];
    const int n = qBase + r + (hh << 3);
    P.O64[((size_t)(m * 2 + b) * NPOS + n) * PDIM + h * HDIM + l16] = (_Float16)val;
  }
}

// ---------------------------------------------------------------------------
// Kernel 3: output projection + bias + residual -> Y (f32), plus per-tile
// GroupNorm partial sums (one 16-channel tile == one GN group).
// ---------------------------------------------------------------------------
__global__ __launch_bounds__(128) void oproj_kernel(Params P) {
  const int lane = threadIdx.x & 31, wv = threadIdx.x >> 5;
  int widx = blockIdx.x * 4 + wv;              // [0, 24576)
  const int m = widx / 8192; widx -= m * 8192;
  const int b = widx >> 12;  widx &= 4095;
  const int ct = widx >> 8;
  const int nt = widx & 255;
  const int hh = lane >> 4, l16 = lane & 15;
  const float* Wo = P.Wo[m];
  const _Float16* Ob = P.O64 + (size_t)(m * 2 + b) * NPOS * PDIM;

  v8f acc = {};
#pragma unroll
  for (int j = 0; j < 2; j++) {
    v16h af;
    const int c = ct * 16 + l16;
#pragma unroll
    for (int e = 0; e < 16; e++) {
      const int p = j * 32 + (hh << 4) + e;
      af[e] = (_Float16)Wo[c * PDIM + p];
    }
    v16h bf = *(const v16h*)(Ob + (size_t)(nt * 16 + l16) * PDIM + j * 32 + (hh << 4));
    acc = WMMA_F16(af, bf, acc);
  }

  const float* feat = P.feat[m] + (size_t)b * CCH * NPOS;
  float s = 0.f, s2 = 0.f;
#pragma unroll
  for (int r = 0; r < 8; r++) {
    const int c = ct * 16 + r + (hh << 3);
    const int n = nt * 16 + l16;
    const float y = acc[r] + P.bo[m][c] + feat[(size_t)c * NPOS + n];
    P.Y[((size_t)(m * 2 + b) * CCH + c) * NPOS + n] = y;
    s += y; s2 += y * y;
  }
#pragma unroll
  for (int off = 1; off < 32; off <<= 1) {
    s  += __shfl_xor(s,  off, 32);
    s2 += __shfl_xor(s2, off, 32);
  }
  if (lane == 0) {
    const int slot = ((m * 2 + b) * 16 + ct) * 256 + nt;
    P.part[slot * 2]     = s;
    P.part[slot * 2 + 1] = s2;
  }
}

// ---------------------------------------------------------------------------
// Kernel 4: reduce partials -> mean / inv-std per (mod,batch,group)
// ---------------------------------------------------------------------------
__global__ __launch_bounds__(256) void stats_kernel(Params P) {
  __shared__ float ss[256], ss2[256];
  const int g = blockIdx.x;                    // [0, 96)
  const int t = threadIdx.x;
  ss[t]  = P.part[(g * 256 + t) * 2];
  ss2[t] = P.part[(g * 256 + t) * 2 + 1];
  __syncthreads();
  for (int st = 128; st > 0; st >>= 1) {
    if (t < st) { ss[t] += ss[t + st]; ss2[t] += ss2[t + st]; }
    __syncthreads();
  }
  if (t == 0) {
    const float inv = 1.0f / 65536.0f;         // 16 channels * 4096 positions
    const float mean = ss[0] * inv;
    const float var  = ss2[0] * inv - mean * mean;
    P.stats[g * 2]     = mean;
    P.stats[g * 2 + 1] = rsqrtf(var + GNEPS);
  }
}

// ---------------------------------------------------------------------------
// Kernel 5: normalize -> d_out (layout matches reference tuple concat)
// ---------------------------------------------------------------------------
__global__ __launch_bounds__(256) void norm_kernel(Params P) {
  const size_t idx = (size_t)blockIdx.x * 256 + threadIdx.x;  // [0, 6291456)
  const int c  = (int)((idx >> 12) & 255);
  const int mb = (int)(idx >> 20);             // (m*2+b)
  const int m  = mb >> 1;
  const int g  = mb * 16 + (c >> 4);
  const float mean = P.stats[g * 2];
  const float is   = P.stats[g * 2 + 1];
  const float y = P.Y[idx];
  P.out[idx] = (y - mean) * is * P.gam[m][c] + P.bet[m][c];
}

// ---------------------------------------------------------------------------
extern "C" void kernel_launch(void* const* d_in, const int* in_sizes, int n_in,
                              void* d_out, int out_size, void* d_ws, size_t ws_size,
                              hipStream_t stream) {
  Params P;
  for (int m = 0; m < 3; m++) {
    P.feat[m] = (const float*)d_in[m];
    const int base = 3 + m * 10;
    P.Wq[m]  = (const float*)d_in[base + 0];
    P.bq[m]  = (const float*)d_in[base + 1];
    P.Wk[m]  = (const float*)d_in[base + 2];
    P.bk[m]  = (const float*)d_in[base + 3];
    P.Wv[m]  = (const float*)d_in[base + 4];
    P.bv[m]  = (const float*)d_in[base + 5];
    P.Wo[m]  = (const float*)d_in[base + 6];
    P.bo[m]  = (const float*)d_in[base + 7];
    P.gam[m] = (const float*)d_in[base + 8];
    P.bet[m] = (const float*)d_in[base + 9];
  }
  char* ws = (char*)d_ws;
  const size_t nQ = (size_t)3 * BATCH * NHEAD * NPOS * HDIM;  // 1,572,864 halves
  P.Qt  = (_Float16*)(ws);
  P.Kt  = (_Float16*)(ws + nQ * 2);
  P.Vd  = (_Float16*)(ws + nQ * 4);
  P.O64 = (_Float16*)(ws + nQ * 6);                           // same elem count
  P.Y   = (float*)(ws + nQ * 8);
  P.part  = (float*)(ws + nQ * 8 + (size_t)3 * BATCH * CCH * NPOS * 4);
  P.stats = (float*)(ws + nQ * 8 + (size_t)3 * BATCH * CCH * NPOS * 4 + 49152 * 4);
  P.out = (float*)d_out;

  proj_kernel <<<1152,  128, 0, stream>>>(P);   // 4608 waves
  attn_kernel <<<1536,  128, 0, stream>>>(P);   // 6144 waves
  oproj_kernel<<<6144,  128, 0, stream>>>(P);   // 24576 waves
  stats_kernel<<<96,    256, 0, stream>>>(P);
  norm_kernel <<<24576, 256, 0, stream>>>(P);
}
